// GNNLayer_87368224735831
// MI455X (gfx1250) — matrix operand order repaired and checked by reading, not compiled, not run on hardware
//
#include <hip/hip_runtime.h>

// GCNConv (x@W, symmetric-norm message passing with self-loops) + BatchNorm1d
// (batch stats) + ReLU for MI455X (gfx1250, wave32).
//
// Phase plan (all on `stream`, graph-capture safe):
//   0) memset: agg(=d_out), deg, BN partial sums
//   1) prep:   Wt[n][k] = bf16(W[k][n])   (32 KB, lives in cache)
//   2) degree: atomicAdd(deg[dst], 1)     ; dinv = rsqrt(deg+1)  (+1 = self loop)
//   3) GEMM:   H = X @ W via v_wmma_f32_16x16x32_bf16 (f32 accum),
//              hardware f32->bf16 converts, non-temporal X reads
//   4) scatter: one wave per edge, 128 dims as float4/lane, f32 atomics into L2
//              (H and agg both fit in the 192 MB L2 -> atomics resolve on-chip)
//   5) self-loop + bias: agg += H*dinv^2 + b (non-atomic, after scatter)
//   6) BN stats (column sums / sumsq via L2 atomics), finalize scale/shift,
//      fused scale+shift+ReLU applied in-place on d_out.

typedef __attribute__((ext_vector_type(16))) __bf16 v16bf;
typedef __attribute__((ext_vector_type(8)))  float  v8f;
typedef __attribute__((ext_vector_type(4)))  float  v4f;   // true vector type
                                                           // (HIP float4 is a
                                                           // struct; builtins
                                                           // reject it)
#define D 128

// ---- 1) W -> Wt (transposed, bf16) -----------------------------------------
__global__ void prep_wt_kernel(const float* __restrict__ W,
                               __bf16* __restrict__ Wt) {
  int i = blockIdx.x * blockDim.x + threadIdx.x;  // 16384 = 128*128
  if (i < D * D) {
    int n = i >> 7, k = i & (D - 1);
    Wt[n * D + k] = (__bf16)W[k * D + n];         // hw cvt, RTNE
  }
}

// ---- 2) degree / dinv ------------------------------------------------------
__global__ void degree_kernel(const int* __restrict__ dst,
                              float* __restrict__ deg, int E) {
  int e = blockIdx.x * blockDim.x + threadIdx.x;
  if (e < E) atomicAdd(&deg[dst[e]], 1.0f);
}

__global__ void dinv_kernel(float* __restrict__ deg, int N) {
  int n = blockIdx.x * blockDim.x + threadIdx.x;
  if (n < N) deg[n] = rsqrtf(deg[n] + 1.0f);  // +1 self-loop; always > 0
}

// ---- 3) H = X @ W via WMMA bf16 -------------------------------------------
// One wave per 16-row M-tile; 8 N-tiles x 4 K-steps = 32 WMMAs per wave.
__global__ void __launch_bounds__(256)
gemm_wmma_kernel(const float* __restrict__ X,
                 const __bf16* __restrict__ Wt,
                 float* __restrict__ H, int mtiles) {
  const int lane  = threadIdx.x & 31;
  const int wave  = threadIdx.x >> 5;
  const int mtile = blockIdx.x * 8 + wave;   // wave-uniform guard: EXEC stays full
  if (mtile >= mtiles) return;
  const int m0   = mtile * 16;
  const int half = lane >> 4;                // 0: lanes 0-15, 1: lanes 16-31
  const int l15  = lane & 15;

  const v8f vz = {0.f, 0.f, 0.f, 0.f, 0.f, 0.f, 0.f, 0.f};
  v8f acc[8];
#pragma unroll
  for (int i = 0; i < 8; ++i) acc[i] = vz;

  const float* xrow = X + (size_t)(m0 + l15) * D;

#pragma unroll
  for (int kk = 0; kk < 4; ++kk) {
    const int k0 = kk * 32;
    // A fragment (16-bit 16x32 layout): lanes 0-15 hold K {k0+0..7, k0+16..23},
    // lanes 16-31 hold K {k0+8..15, k0+24..31}; element pairs are consecutive K.
    const int ka = k0 + half * 8;
    v4f q0 = __builtin_nontemporal_load((const v4f*)(xrow + ka));
    v4f q1 = __builtin_nontemporal_load((const v4f*)(xrow + ka + 4));
    v4f q2 = __builtin_nontemporal_load((const v4f*)(xrow + ka + 16));
    v4f q3 = __builtin_nontemporal_load((const v4f*)(xrow + ka + 20));
    float fa[16] = {q0.x, q0.y, q0.z, q0.w, q1.x, q1.y, q1.z, q1.w,
                    q2.x, q2.y, q2.z, q2.w, q3.x, q3.y, q3.z, q3.w};
    v16bf a;
#pragma unroll
    for (int i = 0; i < 16; ++i) a[i] = (__bf16)fa[i];   // hw cvt (packed)

    // B fragment (16-bit 32x16): lanes 0-15 -> col N=lane, K {k0+0..7, k0+16..23};
    // lanes 16-31 -> col N=lane-16, K {k0+8..15, k0+24..31}. Wt rows are contiguous in K.
    const int kb = k0 + half * 8;
#pragma unroll
    for (int nt = 0; nt < 8; ++nt) {
      const __bf16* bp = Wt + (size_t)(nt * 16 + l15) * D + kb;
      v16bf bfrag;
#pragma unroll
      for (int i = 0; i < 8; ++i) { bfrag[i] = bp[i]; bfrag[i + 8] = bp[i + 16]; }
      acc[nt] = __builtin_amdgcn_wmma_f32_16x16x32_bf16(
          false, a, false, bfrag, (short)0, acc[nt], false, false);
    }
  }

  // C/D layout: VGPR r -> row m0 + 8*half + r, col nt*16 + l15.
#pragma unroll
  for (int nt = 0; nt < 8; ++nt) {
    float* out = H + (size_t)(m0 + half * 8) * D + nt * 16 + l15;
#pragma unroll
    for (int r = 0; r < 8; ++r) out[(size_t)r * D] = acc[nt][r];
  }
}

// ---- 4) edge scatter: one wave per edge ------------------------------------
__global__ void __launch_bounds__(256)
scatter_kernel(const int* __restrict__ src, const int* __restrict__ dst,
               const float* __restrict__ H, const float* __restrict__ dinv,
               float* __restrict__ agg, int E) {
  int e    = (int)((blockIdx.x * blockDim.x + threadIdx.x) >> 5);
  int lane = threadIdx.x & 31;
  if (e >= E) return;
  int s = src[e], d = dst[e];
  float norm = dinv[s] * dinv[d];
  v4f m = ((const v4f*)(H + (size_t)s * D))[lane];
  float* ap = agg + (size_t)d * D + lane * 4;
  atomicAdd(ap + 0, m.x * norm);
  atomicAdd(ap + 1, m.y * norm);
  atomicAdd(ap + 2, m.z * norm);
  atomicAdd(ap + 3, m.w * norm);
}

// ---- 5) self-loop + bias ---------------------------------------------------
__global__ void selfloop_bias_kernel(const float* __restrict__ H,
                                     const float* __restrict__ dinv,
                                     const float* __restrict__ b,
                                     float* __restrict__ agg, int nquads) {
  int i = blockIdx.x * blockDim.x + threadIdx.x;  // N*32 float4's
  if (i >= nquads) return;
  int n = i >> 5, q = i & 31;
  float di = dinv[n];
  float w = di * di;
  v4f h  = ((const v4f*)(H + (size_t)n * D))[q];
  v4f bb = ((const v4f*)b)[q];
  v4f* ap = (v4f*)(agg + (size_t)n * D) + q;
  v4f a = *ap;
  a = a + h * w + bb;           // ext-vector ops: elementwise fma
  *ap = a;
}

// ---- 6) BatchNorm ----------------------------------------------------------
__global__ void __launch_bounds__(256)
bn_stats_kernel(const float* __restrict__ agg, float* __restrict__ gsum,
                float* __restrict__ gsumsq, int N) {
  const int c  = threadIdx.x & (D - 1);
  const int rg = threadIdx.x >> 7;          // two row groups per block
  int r    = blockIdx.x * D + rg;
  int rend = blockIdx.x * D + D;
  if (rend > N) rend = N;
  float s = 0.f, s2 = 0.f;
  for (; r < rend; r += 2) {
    float v = agg[(size_t)r * D + c];
    s += v;
    s2 += v * v;
  }
  atomicAdd(&gsum[c], s);
  atomicAdd(&gsumsq[c], s2);
}

__global__ void bn_finalize_kernel(const float* __restrict__ gsum,
                                   const float* __restrict__ gsumsq,
                                   const float* __restrict__ gamma,
                                   const float* __restrict__ beta,
                                   float* __restrict__ scale,
                                   float* __restrict__ shift, int N) {
  int c = threadIdx.x;
  if (c < D) {
    float inv_n = 1.0f / (float)N;
    float mean = gsum[c] * inv_n;
    float var  = gsumsq[c] * inv_n - mean * mean;  // biased var (matches ref)
    float sc   = gamma[c] * rsqrtf(var + 1e-5f);
    scale[c] = sc;
    shift[c] = beta[c] - mean * sc;
  }
}

__global__ void bn_apply_kernel(const float* __restrict__ scale,
                                const float* __restrict__ shift,
                                float* __restrict__ agg, int nquads) {
  int i = blockIdx.x * blockDim.x + threadIdx.x;  // N*32 float4's, in-place
  if (i >= nquads) return;
  int q = i & 31;
  v4f sc = ((const v4f*)scale)[q];
  v4f sh = ((const v4f*)shift)[q];
  v4f a = ((v4f*)agg)[i];
  a = a * sc + sh;
  a.x = fmaxf(a.x, 0.f);
  a.y = fmaxf(a.y, 0.f);
  a.z = fmaxf(a.z, 0.f);
  a.w = fmaxf(a.w, 0.f);
  ((v4f*)agg)[i] = a;
}

static inline size_t align256(size_t x) { return (x + 255) & ~(size_t)255; }

extern "C" void kernel_launch(void* const* d_in, const int* in_sizes, int n_in,
                              void* d_out, int out_size, void* d_ws, size_t ws_size,
                              hipStream_t stream) {
  const float* X     = (const float*)d_in[0];
  const int*   EI    = (const int*)d_in[1];   // JAX canonicalizes int64->int32
  const float* W     = (const float*)d_in[2];
  const float* bias  = (const float*)d_in[3];
  const float* gamma = (const float*)d_in[4];
  const float* beta  = (const float*)d_in[5];

  const int N = in_sizes[0] / D;      // 100000
  const int E = in_sizes[1] / 2;      // 600000
  const int* src = EI;
  const int* dst = EI + E;
  float* agg = (float*)d_out;         // accumulator lives in d_out

  // workspace carve-out (~51.6 MB)
  char* ws = (char*)d_ws;
  size_t off = 0;
  float* H = (float*)(ws + off);             off += align256((size_t)N * D * sizeof(float));
  float* dinv = (float*)(ws + off);          off += align256((size_t)N * sizeof(float));
  float* gsum = (float*)(ws + off);          off += 512;
  float* gsumsq = (float*)(ws + off);        off += 512;
  float* scale = (float*)(ws + off);         off += 512;
  float* shift = (float*)(ws + off);         off += 512;
  __bf16* Wt = (__bf16*)(ws + off);

  // 0) zero the accumulators (d_out is poisoned by the harness)
  (void)hipMemsetAsync(agg, 0, (size_t)N * D * sizeof(float), stream);
  (void)hipMemsetAsync(dinv, 0, (size_t)N * sizeof(float), stream);
  (void)hipMemsetAsync(gsum, 0, 1024, stream);  // gsum + gsumsq contiguous

  // 1) transpose+convert W
  prep_wt_kernel<<<(D * D + 255) / 256, 256, 0, stream>>>(W, Wt);

  // 2) degree -> dinv
  degree_kernel<<<(E + 255) / 256, 256, 0, stream>>>(dst, dinv, E);
  dinv_kernel<<<(N + 255) / 256, 256, 0, stream>>>(dinv, N);

  // 3) H = X @ W (WMMA)
  const int mtiles = N / 16;  // 6250, exact
  gemm_wmma_kernel<<<(mtiles + 7) / 8, 256, 0, stream>>>(X, Wt, H, mtiles);

  // 4) edge scatter (8 waves per block, one wave per edge)
  scatter_kernel<<<(E + 7) / 8, 256, 0, stream>>>(src, dst, H, dinv, agg, E);

  // 5) self-loops + bias
  const int nquads = N * (D / 4);
  selfloop_bias_kernel<<<(nquads + 255) / 256, 256, 0, stream>>>(H, dinv, bias, agg, nquads);

  // 6) BatchNorm + ReLU
  bn_stats_kernel<<<(N + D - 1) / D, 256, 0, stream>>>(agg, gsum, gsumsq, N);
  bn_finalize_kernel<<<1, 128, 0, stream>>>(gsum, gsumsq, gamma, beta, scale, shift, N);
  bn_apply_kernel<<<(nquads + 255) / 256, 256, 0, stream>>>(scale, shift, agg, nquads);
}
